// Block_14688788152954
// MI455X (gfx1250) — matrix operand order repaired and checked by reading, not compiled, not run on hardware
//
#include <hip/hip_runtime.h>
#include <hip/hip_bf16.h>
#include <math.h>

// ---------------- problem constants ----------------
constexpr int B = 2, T = 2048, C = 2048, F = 8192;
constexpr int M = B * T;           // 4096 rows through all GEMMs
constexpr float LN_EPS = 1e-5f;

// ---------------- WMMA types ----------------
typedef __attribute__((ext_vector_type(16))) _Float16 v16h;
typedef __attribute__((ext_vector_type(8)))  float    v8f;

// ---------------- GEMM tiling ----------------
#define TM 256     // block M tile (8 waves x 32 rows)
#define TN 64      // block N tile
#define TK 32      // K step (one WMMA K)
#define LDA_S 40   // padded LDS stride (halves); 80B rows stay 16B aligned
#define LDB_S 40

// =====================================================================
// Weight transpose + f32->f16 convert:  in[K][N] (f32) -> out[N][K] (f16)
// =====================================================================
__global__ __launch_bounds__(256) void transpose_cvt(const float* __restrict__ in,
                                                     _Float16* __restrict__ out,
                                                     int K, int N) {
  __shared__ float tile[32][33];
  const int nb = blockIdx.x * 32, kb = blockIdx.y * 32;
  const int tx = threadIdx.x, ty = threadIdx.y;   // block (32,8)
  for (int i = ty; i < 32; i += 8)
    tile[i][tx] = in[(size_t)(kb + i) * N + nb + tx];
  __syncthreads();
  for (int i = ty; i < 32; i += 8)
    out[(size_t)(nb + i) * K + kb + tx] = (_Float16)tile[tx][i];
}

// =====================================================================
// WMMA GEMM:  out = epilogue( A[M][K](f16) * Bt[N][K](f16)^T )
// MODE 0: f32 identity   MODE 1: f16 relu(x)^2   MODE 2: f32 sigmoid
// NN/KK compile-time so epilogue/loop offsets fold into immediates.
// Software-pipelined, double-buffered LDS, manual 2x unroll so the
// buffer index is a literal (no per-iter LDS address math).
// =====================================================================
template <int MODE, int NN, int KK>
__global__ __launch_bounds__(256) void gemm_wmma(const _Float16* __restrict__ A,
                                                 const _Float16* __restrict__ Bt,
                                                 float* __restrict__ outF,
                                                 _Float16* __restrict__ outH) {
  __shared__ alignas(16) _Float16 As[2][TM * LDA_S];
  __shared__ alignas(16) _Float16 Bs[2][TN * LDB_S];

  const int tid  = threadIdx.x;
  const int wave = tid >> 5;
  const int lane = tid & 31;
  const int lm   = lane & 15;      // M (A) / N (B,D) within fragment
  const int hg   = lane >> 4;      // half-group select
  const int m0   = blockIdx.y * TM;
  const int n0   = blockIdx.x * TN;

  // global->LDS staging assignments
  // A: each thread owns one row (tid) of the 256-row tile: 32 halves = 4x b128
  // B: 64 rows, 4 threads per row: 8 halves = 1x b128 each
  const int br = tid >> 2;
  const int bk = (tid & 3) * 8;

  const _Float16* gA_row = A + (size_t)(m0 + tid) * KK;
  const _Float16* gB_row = Bt + (size_t)(n0 + br) * KK + bk;

  v8f acc[2][4];
  for (int s = 0; s < 2; ++s)
    for (int j = 0; j < 4; ++j)
      for (int e = 0; e < 8; ++e) acc[s][j][e] = 0.f;

  struct Staged { uint4 a0, a1, a2, a3, b; };
  auto gload = [&](int kofs) -> Staged {
    Staged st;
    const uint4* ga = reinterpret_cast<const uint4*>(gA_row + kofs);
    st.a0 = ga[0]; st.a1 = ga[1]; st.a2 = ga[2]; st.a3 = ga[3];
    st.b = *reinterpret_cast<const uint4*>(gB_row + kofs);
    if (kofs + TK < KK) {   // prefetch tile after next -> global_prefetch_b8
      __builtin_prefetch((const void*)(gA_row + kofs + TK), 0, 1);
      __builtin_prefetch((const void*)(gB_row + kofs + TK), 0, 1);
    }
    return st;
  };
  auto sstore = [&](int buf, const Staged& st) {
    uint4* sa = reinterpret_cast<uint4*>(&As[buf][tid * LDA_S]);
    sa[0] = st.a0; sa[1] = st.a1; sa[2] = st.a2; sa[3] = st.a3;
    *reinterpret_cast<uint4*>(&Bs[buf][br * LDB_S + bk]) = st.b;
  };
  auto compute = [&](int buf) {
    // A fragments: lane holds M=lm; half h => K = 16*hg + h
    v16h a0, a1, b0, b1, b2, b3;
    const _Float16* pa = &As[buf][(wave * 32 + lm) * LDA_S + hg * 16];
    reinterpret_cast<uint4*>(&a0)[0] = *reinterpret_cast<const uint4*>(pa);
    reinterpret_cast<uint4*>(&a0)[1] = *reinterpret_cast<const uint4*>(pa + 8);
    reinterpret_cast<uint4*>(&a1)[0] = *reinterpret_cast<const uint4*>(pa + 16 * LDA_S);
    reinterpret_cast<uint4*>(&a1)[1] = *reinterpret_cast<const uint4*>(pa + 16 * LDA_S + 8);
    // B fragments: lane holds N=lm; half h => K = 16*hg + h (K-contiguous in Bs)
    const _Float16* pb = &Bs[buf][lm * LDB_S + hg * 16];
    reinterpret_cast<uint4*>(&b0)[0] = reinterpret_cast<const uint4*>(pb)[0];
    reinterpret_cast<uint4*>(&b0)[1] = reinterpret_cast<const uint4*>(pb)[1];
    reinterpret_cast<uint4*>(&b1)[0] = reinterpret_cast<const uint4*>(pb + 16 * LDB_S)[0];
    reinterpret_cast<uint4*>(&b1)[1] = reinterpret_cast<const uint4*>(pb + 16 * LDB_S)[1];
    reinterpret_cast<uint4*>(&b2)[0] = reinterpret_cast<const uint4*>(pb + 32 * LDB_S)[0];
    reinterpret_cast<uint4*>(&b2)[1] = reinterpret_cast<const uint4*>(pb + 32 * LDB_S)[1];
    reinterpret_cast<uint4*>(&b3)[0] = reinterpret_cast<const uint4*>(pb + 48 * LDB_S)[0];
    reinterpret_cast<uint4*>(&b3)[1] = reinterpret_cast<const uint4*>(pb + 48 * LDB_S)[1];
    acc[0][0] = __builtin_amdgcn_wmma_f32_16x16x32_f16(false, a0, false, b0, (short)0, acc[0][0], false, false);
    acc[1][0] = __builtin_amdgcn_wmma_f32_16x16x32_f16(false, a1, false, b0, (short)0, acc[1][0], false, false);
    acc[0][1] = __builtin_amdgcn_wmma_f32_16x16x32_f16(false, a0, false, b1, (short)0, acc[0][1], false, false);
    acc[1][1] = __builtin_amdgcn_wmma_f32_16x16x32_f16(false, a1, false, b1, (short)0, acc[1][1], false, false);
    acc[0][2] = __builtin_amdgcn_wmma_f32_16x16x32_f16(false, a0, false, b2, (short)0, acc[0][2], false, false);
    acc[1][2] = __builtin_amdgcn_wmma_f32_16x16x32_f16(false, a1, false, b2, (short)0, acc[1][2], false, false);
    acc[0][3] = __builtin_amdgcn_wmma_f32_16x16x32_f16(false, a0, false, b3, (short)0, acc[0][3], false, false);
    acc[1][3] = __builtin_amdgcn_wmma_f32_16x16x32_f16(false, a1, false, b3, (short)0, acc[1][3], false, false);
  };

  // ---- prologue: stage tile k0=0 into buffer 0 ----
  sstore(0, gload(0));
  __syncthreads();

  // KK/TK is even for all instantiations: pair the steps so buffer
  // indices are literals.
#pragma unroll 1
  for (int k0 = 0; k0 < KK; k0 += 2 * TK) {
    {  // step A: compute buf0; stage k0+TK (always in range) -> buf1
      Staged st = gload(k0 + TK);
      compute(0);
      sstore(1, st);
      __syncthreads();
    }
    {  // step B: compute buf1; stage k0+2*TK -> buf0 (if any)
      const bool hn = (k0 + 2 * TK) < KK;
      Staged st;
      if (hn) st = gload(k0 + 2 * TK);
      compute(1);
      if (hn) sstore(0, st);
      __syncthreads();
    }
  }

  // D layout: element e => M = e + 8*hg ; N = lm (per 16x16 subtile j)
  for (int s = 0; s < 2; ++s) {
    for (int j = 0; j < 4; ++j) {
      const int nn = n0 + j * 16 + lm;
      const int mbase = m0 + wave * 32 + s * 16 + 8 * hg;
      if (MODE == 1) {
        _Float16* po = outH + (size_t)mbase * NN + nn;
#pragma unroll
        for (int e = 0; e < 8; ++e) {
          const float v = acc[s][j][e];
          const float r = v > 0.f ? v : 0.f;
          po[e * NN] = (_Float16)(r * r);
        }
      } else {
        float* po = outF + (size_t)mbase * NN + nn;
#pragma unroll
        for (int e = 0; e < 8; ++e) {
          const float v = acc[s][j][e];
          po[e * NN] = (MODE == 0) ? v : 1.f / (1.f + expf(-v));
        }
      }
    }
  }
}

// =====================================================================
// block-wide f32 sum (256 threads)
// =====================================================================
__device__ __forceinline__ float block_sum(float v, float* sbuf) {
  const int tid = threadIdx.x;
  sbuf[tid] = v;
  __syncthreads();
  for (int s = 128; s > 0; s >>= 1) {
    if (tid < s) sbuf[tid] += sbuf[tid + s];
    __syncthreads();
  }
  const float r = sbuf[0];
  __syncthreads();
  return r;
}

// =====================================================================
// prep_att: LN1 + time-shift mix -> f16 xk/xv/xr ; writes x1[:, -1]
// (each block recomputes LN of the previous row -> no cross-block dep)
// =====================================================================
__global__ __launch_bounds__(256) void prep_att(
    const float* __restrict__ x, const float* __restrict__ att_shift,
    const float* __restrict__ g1, const float* __restrict__ b1,
    const float* __restrict__ tmk, const float* __restrict__ tmv,
    const float* __restrict__ tmr,
    _Float16* __restrict__ xk, _Float16* __restrict__ xv, _Float16* __restrict__ xr,
    float* __restrict__ x1_last) {
  __shared__ float sbuf[256];
  const int tid = threadIdx.x;
  const int row = blockIdx.x;
  const int b = row / T, t = row % T;
  const float* xt = x + (size_t)row * C;
  const float* xp = xt - C;

  float s = 0.f, ss = 0.f;
  for (int c = tid; c < C; c += 256) { float v = xt[c]; s += v; ss += v * v; }
  s = block_sum(s, sbuf); ss = block_sum(ss, sbuf);
  const float mt = s * (1.f / C);
  const float rt = rsqrtf(ss * (1.f / C) - mt * mt + LN_EPS);

  float mp = 0.f, rp = 0.f;
  if (t > 0) {
    float s2 = 0.f, ss2 = 0.f;
    for (int c = tid; c < C; c += 256) { float v = xp[c]; s2 += v; ss2 += v * v; }
    s2 = block_sum(s2, sbuf); ss2 = block_sum(ss2, sbuf);
    mp = s2 * (1.f / C);
    rp = rsqrtf(ss2 * (1.f / C) - mp * mp + LN_EPS);
  }

  for (int c = tid; c < C; c += 256) {
    const float x1 = (xt[c] - mt) * rt * g1[c] + b1[c];
    const float xx = (t > 0) ? (xp[c] - mp) * rp * g1[c] + b1[c]
                             : att_shift[(size_t)b * C + c];
    const float k_ = tmk[c], v_ = tmv[c], r_ = tmr[c];
    const size_t o = (size_t)row * C + c;
    xk[o] = (_Float16)(x1 * k_ + xx * (1.f - k_));
    xv[o] = (_Float16)(x1 * v_ + xx * (1.f - v_));
    xr[o] = (_Float16)(x1 * r_ + xx * (1.f - r_));
    if (t == T - 1) x1_last[(size_t)b * C + c] = x1;
  }
}

// =====================================================================
// wkv scan: one thread per (b,c) channel, sequential over T.
// Fuses g = sigmoid(r) * y (f16 out) ; writes final wkv state.
// =====================================================================
__global__ __launch_bounds__(256) void wkv_scan(
    const float* __restrict__ k, const float* __restrict__ v,
    const float* __restrict__ r,
    const float* __restrict__ time_decay, const float* __restrict__ time_first,
    const float* __restrict__ state_in,
    _Float16* __restrict__ g, float* __restrict__ wkv_out) {
  const int idx = blockIdx.x * 256 + threadIdx.x;   // exactly B*C threads
  const int c = idx % C;
  const int b = idx / C;
  const float w = -expf(time_decay[c]);
  const float u = time_first[c];
  float aa = state_in[((size_t)b * 3 + 0) * C + c];
  float bb = state_in[((size_t)b * 3 + 1) * C + c];
  float pp = state_in[((size_t)b * 3 + 2) * C + c];
  size_t base = (size_t)b * T * C + c;
  for (int t = 0; t < T; ++t, base += C) {
    const float kt = k[base], vt = v[base];
    const float ww = u + kt;
    const float p = fmaxf(pp, ww);
    const float e1 = expf(pp - p), e2 = expf(ww - p);
    const float y = (e1 * aa + e2 * vt) / (e1 * bb + e2);
    const float ww2 = pp + w;
    const float p2 = fmaxf(ww2, kt);
    const float e1b = expf(ww2 - p2), e2b = expf(kt - p2);
    aa = e1b * aa + e2b * vt;
    bb = e1b * bb + e2b;
    pp = p2;
    const float sg = 1.f / (1.f + expf(-r[base]));
    g[base] = (_Float16)(sg * y);
  }
  wkv_out[((size_t)b * 3 + 0) * C + c] = aa;
  wkv_out[((size_t)b * 3 + 1) * C + c] = bb;
  wkv_out[((size_t)b * 3 + 2) * C + c] = pp;
}

// =====================================================================
// prep_ffn: xatt = x + woz ; LN2 + ffn time-shift mix -> f16 xk2/xr2
// stores xatt (needed for final residual) and x2[:, -1]
// =====================================================================
__global__ __launch_bounds__(256) void prep_ffn(
    const float* __restrict__ x, const float* __restrict__ woz,
    const float* __restrict__ ffn_shift,
    const float* __restrict__ g2, const float* __restrict__ b2,
    const float* __restrict__ fmk, const float* __restrict__ fmr,
    float* __restrict__ xatt,
    _Float16* __restrict__ xk2, _Float16* __restrict__ xr2,
    float* __restrict__ x2_last) {
  __shared__ float sbuf[256];
  const int tid = threadIdx.x;
  const int row = blockIdx.x;
  const int b = row / T, t = row % T;
  const float* xt = x + (size_t)row * C;
  const float* zt = woz + (size_t)row * C;

  float s = 0.f, ss = 0.f;
  for (int c = tid; c < C; c += 256) {
    const float xa = xt[c] + zt[c];
    xatt[(size_t)row * C + c] = xa;
    s += xa; ss += xa * xa;
  }
  s = block_sum(s, sbuf); ss = block_sum(ss, sbuf);
  const float mt = s * (1.f / C);
  const float rt = rsqrtf(ss * (1.f / C) - mt * mt + LN_EPS);

  float mp = 0.f, rp = 0.f;
  if (t > 0) {
    float s2 = 0.f, ss2 = 0.f;
    for (int c = tid; c < C; c += 256) {
      const float xa = xt[c - C] + zt[c - C];
      s2 += xa; ss2 += xa * xa;
    }
    s2 = block_sum(s2, sbuf); ss2 = block_sum(ss2, sbuf);
    mp = s2 * (1.f / C);
    rp = rsqrtf(ss2 * (1.f / C) - mp * mp + LN_EPS);
  }

  for (int c = tid; c < C; c += 256) {
    const float xa = xt[c] + zt[c];
    const float x2 = (xa - mt) * rt * g2[c] + b2[c];
    const float xx = (t > 0) ? ((xt[c - C] + zt[c - C]) - mp) * rp * g2[c] + b2[c]
                             : ffn_shift[(size_t)b * C + c];
    const float k_ = fmk[c], r_ = fmr[c];
    const size_t o = (size_t)row * C + c;
    xk2[o] = (_Float16)(x2 * k_ + xx * (1.f - k_));
    xr2[o] = (_Float16)(x2 * r_ + xx * (1.f - r_));
    if (t == T - 1) x2_last[(size_t)b * C + c] = x2;
  }
}

// =====================================================================
// final: out = xatt + gate * kv
// =====================================================================
__global__ __launch_bounds__(256) void final_add(const float* __restrict__ xatt,
                                                 const float* __restrict__ gate,
                                                 const float* __restrict__ kv,
                                                 float* __restrict__ out, size_t n) {
  for (size_t i = (size_t)blockIdx.x * 256 + threadIdx.x; i < n;
       i += (size_t)gridDim.x * 256)
    out[i] = xatt[i] + gate[i] * kv[i];
}

// =====================================================================
// launch
// =====================================================================
extern "C" void kernel_launch(void* const* d_in, const int* in_sizes, int n_in,
                              void* d_out, int out_size, void* d_ws, size_t ws_size,
                              hipStream_t stream) {
  (void)in_sizes; (void)n_in; (void)out_size; (void)ws_size;

  const float* x          = (const float*)d_in[0];
  const float* att_shift  = (const float*)d_in[1];
  const float* wkv_state  = (const float*)d_in[2];
  const float* ffn_shift  = (const float*)d_in[3];
  const float* ln1_g      = (const float*)d_in[4];
  const float* ln1_b      = (const float*)d_in[5];
  const float* ln2_g      = (const float*)d_in[6];
  const float* ln2_b      = (const float*)d_in[7];
  const float* tmk        = (const float*)d_in[8];
  const float* tmv        = (const float*)d_in[9];
  const float* tmr        = (const float*)d_in[10];
  const float* time_decay = (const float*)d_in[11];
  const float* time_first = (const float*)d_in[12];
  const float* Wk         = (const float*)d_in[13];
  const float* Wv         = (const float*)d_in[14];
  const float* Wr         = (const float*)d_in[15];
  const float* Wo         = (const float*)d_in[16];
  const float* fmk        = (const float*)d_in[17];
  const float* fmr        = (const float*)d_in[18];
  const float* Fk         = (const float*)d_in[19];
  const float* Fr         = (const float*)d_in[20];
  const float* Fv         = (const float*)d_in[21];

  float* out      = (float*)d_out;
  float* out_x    = out;                                    // (B,T,C)
  float* out_x1l  = out + (size_t)B * T * C;                // (B,C)
  float* out_wkv  = out_x1l + (size_t)B * C;                // (B,3,C)
  float* out_x2l  = out_wkv + (size_t)B * 3 * C;            // (B,C)

  // bump allocator over workspace
  char* wp = (char*)d_ws;
  auto alloc = [&](size_t bytes) -> void* {
    void* p = (void*)wp;
    wp += (bytes + 255) & ~(size_t)255;
    return p;
  };
  _Float16* Wkt = (_Float16*)alloc((size_t)C * C * 2);
  _Float16* Wvt = (_Float16*)alloc((size_t)C * C * 2);
  _Float16* Wrt = (_Float16*)alloc((size_t)C * C * 2);
  _Float16* Wot = (_Float16*)alloc((size_t)C * C * 2);
  _Float16* Frt = (_Float16*)alloc((size_t)C * C * 2);
  _Float16* Fkt = (_Float16*)alloc((size_t)F * C * 2);      // [F][C]
  _Float16* Fvt = (_Float16*)alloc((size_t)C * F * 2);      // [C][F]
  _Float16* xk  = (_Float16*)alloc((size_t)M * C * 2);
  _Float16* xv  = (_Float16*)alloc((size_t)M * C * 2);
  _Float16* xr  = (_Float16*)alloc((size_t)M * C * 2);
  float*    kb  = (float*)alloc((size_t)M * C * 4);
  float*    vb  = (float*)alloc((size_t)M * C * 4);
  float*    rb  = (float*)alloc((size_t)M * C * 4);
  _Float16* gb  = (_Float16*)alloc((size_t)M * C * 2);
  float*    woz = (float*)alloc((size_t)M * C * 4);
  float*    xat = (float*)alloc((size_t)M * C * 4);
  _Float16* xk2 = (_Float16*)alloc((size_t)M * C * 2);
  _Float16* xr2 = (_Float16*)alloc((size_t)M * C * 2);
  _Float16* kf  = (_Float16*)alloc((size_t)M * F * 2);
  float*    kvb = (float*)alloc((size_t)M * C * 4);
  float*    frz = (float*)alloc((size_t)M * C * 4);

  const dim3 tb(32, 8);
  // weights -> f16 [N][K]
  transpose_cvt<<<dim3(C / 32, C / 32), tb, 0, stream>>>(Wk, Wkt, C, C);
  transpose_cvt<<<dim3(C / 32, C / 32), tb, 0, stream>>>(Wv, Wvt, C, C);
  transpose_cvt<<<dim3(C / 32, C / 32), tb, 0, stream>>>(Wr, Wrt, C, C);
  transpose_cvt<<<dim3(C / 32, C / 32), tb, 0, stream>>>(Wo, Wot, C, C);
  transpose_cvt<<<dim3(C / 32, C / 32), tb, 0, stream>>>(Fr, Frt, C, C);
  transpose_cvt<<<dim3(F / 32, C / 32), tb, 0, stream>>>(Fk, Fkt, C, F);
  transpose_cvt<<<dim3(C / 32, F / 32), tb, 0, stream>>>(Fv, Fvt, F, C);

  // attention path
  prep_att<<<M, 256, 0, stream>>>(x, att_shift, ln1_g, ln1_b, tmk, tmv, tmr,
                                  xk, xv, xr, out_x1l);
  gemm_wmma<0, C, C><<<dim3(C / TN, M / TM), 256, 0, stream>>>(xk, Wkt, kb, nullptr);
  gemm_wmma<0, C, C><<<dim3(C / TN, M / TM), 256, 0, stream>>>(xv, Wvt, vb, nullptr);
  gemm_wmma<0, C, C><<<dim3(C / TN, M / TM), 256, 0, stream>>>(xr, Wrt, rb, nullptr);
  wkv_scan<<<(B * C) / 256, 256, 0, stream>>>(kb, vb, rb, time_decay, time_first,
                                              wkv_state, gb, out_wkv);
  gemm_wmma<0, C, C><<<dim3(C / TN, M / TM), 256, 0, stream>>>(gb, Wot, woz, nullptr);

  // ffn path
  prep_ffn<<<M, 256, 0, stream>>>(x, woz, ffn_shift, ln2_g, ln2_b, fmk, fmr,
                                  xat, xk2, xr2, out_x2l);
  gemm_wmma<1, F, C><<<dim3(F / TN, M / TM), 256, 0, stream>>>(xk2, Fkt, nullptr, kf);
  gemm_wmma<2, C, C><<<dim3(C / TN, M / TM), 256, 0, stream>>>(xr2, Frt, frz, nullptr);
  gemm_wmma<0, C, F><<<dim3(C / TN, M / TM), 256, 0, stream>>>(kf, Fvt, kvb, nullptr);

  final_add<<<4096, 256, 0, stream>>>(xat, frz, kvb, out_x, (size_t)M * C);
}
